// FieldPredictionNetwork_8126078124545
// MI455X (gfx1250) — matrix operand order repaired and checked by reading, not compiled, not run on hardware
//
#include <hip/hip_runtime.h>
#include <hip/hip_bf16.h>

typedef __attribute__((ext_vector_type(16))) _Float16 v16h;
typedef __attribute__((ext_vector_type(8)))  _Float16 v8h;
typedef __attribute__((ext_vector_type(8)))  float    v8f;
typedef unsigned int u32x4 __attribute__((ext_vector_type(4)));
typedef int          i32x8 __attribute__((ext_vector_type(8)));
typedef int          i32x4 __attribute__((ext_vector_type(4)));

#define TWO_PI 6.2831853071795864f

// ---------------------------------------------------------------------------
// f32 -> f16 conversion
// ---------------------------------------------------------------------------
__global__ void cvt_f32_f16(const float* __restrict__ in, _Float16* __restrict__ out, int n) {
    int i = blockIdx.x * blockDim.x + threadIdx.x;
    if (i < n) out[i] = (_Float16)in[i];
}

// ---------------------------------------------------------------------------
// TDM: DMA a 2D tile of f16 (tile_d1 rows x tile_d0 elems) from a row-major
// [tensor_d1 x tensor_d0] global tensor (row stride stride0 elems) into LDS.
// D# layout per CDNA5 ISA ch.8 (group0 128b, group1 256b, groups2/3 zero: 2D).
// 6-arg builtin form: (g0, g1, g2, g3, g_extra, cpol).
// ---------------------------------------------------------------------------
__device__ __forceinline__ void tdm_load_2d(unsigned lds_byte_addr,
                                            const void* gaddr,
                                            unsigned tensor_d0, unsigned tensor_d1,
                                            unsigned tile_d0,   unsigned tile_d1,
                                            unsigned long long stride0)
{
    unsigned long long ga = (unsigned long long)(uintptr_t)gaddr;
    u32x4 g0;
    g0.x = 1u;                                                   // count=1, user D#
    g0.y = lds_byte_addr;                                        // lds_addr
    g0.z = (unsigned)ga;                                         // global_addr[31:0]
    g0.w = (unsigned)((ga >> 32) & 0x01FFFFFFu) | (2u << 30);    // ga[56:32] | type=2
    i32x8 g1;
    g1[0] = (int)(1u << 16);                                     // data_size = 2 bytes
    g1[1] = (int)((tensor_d0 & 0xFFFFu) << 16);                  // tensor_dim0 lo
    g1[2] = (int)((tensor_d0 >> 16) | ((tensor_d1 & 0xFFFFu) << 16));
    g1[3] = (int)((tensor_d1 >> 16) | ((tile_d0 & 0xFFFFu) << 16)); // tile_dim0
    g1[4] = (int)(tile_d1 & 0xFFFFu);                            // tile_dim1, tile_dim2=0
    g1[5] = (int)(unsigned)(stride0 & 0xFFFFFFFFull);            // dim0_stride lo
    g1[6] = (int)(unsigned)((stride0 >> 32) & 0xFFFFull);        // dim0_stride hi
    g1[7] = 0;
    i32x4 gz4 = (i32x4)0;
    i32x8 gz8 = (i32x8)0;
    __builtin_amdgcn_tensor_load_to_lds(g0, g1, gz4, gz4, gz8, 0);
}

// ---------------------------------------------------------------------------
// WMMA f16 GEMM:  C[M,N] = act(A[M,K] @ W[N,K]^T + bias)
// Wave tile 32x64 (2x4 wmma 16x16); block = WM x WN waves (BM=32*WM, BN=64*WN).
// Operand staging: TDM tensor_load_to_lds, double buffered, TENSORcnt-synced.
// ---------------------------------------------------------------------------
template<int WM, int WN, bool RELU, bool OUT16>
__global__ __launch_bounds__(WM * WN * 32)
void gemm_f16(const _Float16* __restrict__ A,   // [M,K] row major
              const _Float16* __restrict__ W,   // [N,K] row major
              const float*    __restrict__ bias,// [N]
              void*           __restrict__ Cout,// [M,N] f16 or f32
              int M, int N, int K)
{
    constexpr int BM = 32 * WM;
    constexpr int BN = 64 * WN;
    constexpr int BK = 64;

    __shared__ __align__(16) _Float16 As[2][BM][BK];
    __shared__ __align__(16) _Float16 Ws[2][BN][BK];

    const int tid   = threadIdx.x;
    const int wave  = tid / 32;
    const int lane  = tid % 32;
    const int waveM = wave % WM;
    const int waveN = wave / WM;
    const int blockM = blockIdx.x * BM;
    const int blockN = blockIdx.y * BN;
    const int lhalf  = (lane < 16) ? 0 : 1;

    v8f acc[2][4];
    #pragma unroll
    for (int i = 0; i < 2; ++i)
        #pragma unroll
        for (int j = 0; j < 4; ++j) acc[i][j] = (v8f)0.0f;

    const int nIter = K / BK;

    // stage 0 into buffer 0 (wave 0 issues both tile DMAs)
    if (wave == 0) {
        tdm_load_2d((unsigned)(uintptr_t)&As[0][0][0],
                    A + (size_t)blockM * K, (unsigned)K, (unsigned)M,
                    BK, BM, (unsigned long long)K);
        tdm_load_2d((unsigned)(uintptr_t)&Ws[0][0][0],
                    W + (size_t)blockN * K, (unsigned)K, (unsigned)N,
                    BK, BN, (unsigned long long)K);
    }

    for (int it = 0; it < nIter; ++it) {
        const int cur = it & 1;
        if (wave == 0) {
            if (it + 1 < nIter) {
                const int k0n = (it + 1) * BK;
                tdm_load_2d((unsigned)(uintptr_t)&As[cur ^ 1][0][0],
                            A + (size_t)blockM * K + k0n, (unsigned)K, (unsigned)M,
                            BK, BM, (unsigned long long)K);
                tdm_load_2d((unsigned)(uintptr_t)&Ws[cur ^ 1][0][0],
                            W + (size_t)blockN * K + k0n, (unsigned)K, (unsigned)N,
                            BK, BN, (unsigned long long)K);
                // in-order TENSORcnt: <=2 left  => current buffer's 2 DMAs done
                __builtin_amdgcn_s_wait_tensorcnt((short)2);
            } else {
                __builtin_amdgcn_s_wait_tensorcnt((short)0);
            }
        }
        __syncthreads();   // current buffer visible to all waves

        #pragma unroll
        for (int kk = 0; kk < BK; kk += 32) {
            // A fragments: 16x32 f16; lane row M=lane%16; K chunks at kb, kb+16
            v16h afrag[2];
            #pragma unroll
            for (int i = 0; i < 2; ++i) {
                int row = waveM * 32 + i * 16 + (lane & 15);
                int kb  = kk + lhalf * 8;
                v8h lo = *(const v8h*)&As[cur][row][kb];
                v8h hi = *(const v8h*)&As[cur][row][kb + 16];
                afrag[i] = __builtin_shufflevector(lo, hi,
                    0,1,2,3,4,5,6,7,8,9,10,11,12,13,14,15);
            }
            // B fragments: 32x16 f16; lane col N=lane%16; K chunks at kb, kb+16
            v16h bfrag[4];
            #pragma unroll
            for (int j = 0; j < 4; ++j) {
                int col = waveN * 64 + j * 16 + (lane & 15);
                int kb  = kk + lhalf * 8;
                v8h lo = *(const v8h*)&Ws[cur][col][kb];
                v8h hi = *(const v8h*)&Ws[cur][col][kb + 16];
                bfrag[j] = __builtin_shufflevector(lo, hi,
                    0,1,2,3,4,5,6,7,8,9,10,11,12,13,14,15);
            }
            #pragma unroll
            for (int i = 0; i < 2; ++i)
                #pragma unroll
                for (int j = 0; j < 4; ++j)
                    acc[i][j] = __builtin_amdgcn_wmma_f32_16x16x32_f16(
                        false, afrag[i], false, bfrag[j],
                        (short)0, acc[i][j], false, false);
        }
        __syncthreads();   // done reading: buffer may be refilled next iter
    }

    // epilogue: bias + optional relu, f16 or f32 store
    #pragma unroll
    for (int i = 0; i < 2; ++i) {
        #pragma unroll
        for (int j = 0; j < 4; ++j) {
            int n  = blockN + waveN * 64 + j * 16 + (lane & 15);
            float bv = bias[n];
            int mbase = blockM + waveM * 32 + i * 16 + lhalf * 8;
            #pragma unroll
            for (int r = 0; r < 8; ++r) {
                float v = acc[i][j][r] + bv;
                if (RELU) v = fmaxf(v, 0.0f);
                size_t o = (size_t)(mbase + r) * N + n;
                if constexpr (OUT16) ((_Float16*)Cout)[o] = (_Float16)v;
                else                 ((float*)Cout)[o]    = v;
            }
        }
    }
}

// ---------------------------------------------------------------------------
// Width-axis 64-pt DFT of real rows -> complex planes
// ---------------------------------------------------------------------------
__global__ __launch_bounds__(256)
void fft_width64(const float* __restrict__ x, float* __restrict__ yre,
                 float* __restrict__ yim)
{
    __shared__ float row[4][64];
    __shared__ float tc[64], ts[64];
    int t = threadIdx.x;
    if (t < 64) { float a = -TWO_PI * (float)t / 64.0f; __sincosf(a, &ts[t], &tc[t]); }
    int b0 = blockIdx.x * 4;
    int ty = t >> 6, tx = t & 63;
    row[ty][tx] = x[(size_t)(b0 + ty) * 64 + tx];
    __syncthreads();
    float sr = 0.f, si = 0.f;
    #pragma unroll 8
    for (int w = 0; w < 64; ++w) {
        int idx = (w * tx) & 63;
        float v = row[ty][w];
        sr += v * tc[idx];
        si += v * ts[idx];
    }
    size_t o = (size_t)(b0 + ty) * 64 + tx;
    yre[o] = sr; yim[o] = si;
}

// ---------------------------------------------------------------------------
// Batch-axis 32768-pt FFT, four-step: 32768 = N1(256) x N2(128).
// b = 128*n1 + n2; X[k1 + 256*k2] = FFT_128_n2( W_N^{n2 k1} FFT_256_n1(x) )
// LDS-resident radix-2 sub-FFTs, 16 columns per workgroup.
// ---------------------------------------------------------------------------
#define COLS 16

template<int L>
__device__ inline void lds_fft(float (*sre)[COLS + 1], float (*sim)[COLS + 1], float sign)
{
    const int t = threadIdx.x;
    for (int s = 1; (1 << s) <= L; ++s) {
        int m = 1 << s, h = m >> 1;
        for (int idx = t; idx < (L / 2) * COLS; idx += (int)blockDim.x) {
            int bf = idx / COLS, c = idx % COLS;
            int grp = bf / h, pos = bf - grp * h;
            int i0 = grp * m + pos, i1 = i0 + h;
            float ang = sign * TWO_PI * (float)pos / (float)m;
            float sw, cw; __sincosf(ang, &sw, &cw);
            float xr = sre[i1][c], xi = sim[i1][c];
            float tr = cw * xr - sw * xi;
            float ti = cw * xi + sw * xr;
            float ur = sre[i0][c], ui = sim[i0][c];
            sre[i0][c] = ur + tr; sim[i0][c] = ui + ti;
            sre[i1][c] = ur - tr; sim[i1][c] = ui - ti;
        }
        __syncthreads();
    }
}

template<int SIGN>
__global__ __launch_bounds__(256)
void fft_col_stage1(const float* __restrict__ xre, const float* __restrict__ xim,
                    float* __restrict__ tre, float* __restrict__ tim)
{
    constexpr int L = 256, N2 = 128;
    __shared__ float sre[L][COLS + 1];
    __shared__ float sim[L][COLS + 1];
    const int n2 = blockIdx.x;
    const int cbase = blockIdx.y * COLS;
    const int t = threadIdx.x;
    for (int idx = t; idx < L * COLS; idx += 256) {
        int n1 = idx / COLS, c = idx % COLS;
        int r = (int)(__brev((unsigned)n1) >> 24);   // 8-bit reverse
        size_t src = (size_t)(N2 * n1 + n2) * 64 + cbase + c;
        sre[r][c] = xre[src];
        sim[r][c] = xim[src];
    }
    __syncthreads();
    lds_fft<L>(sre, sim, (float)SIGN);
    for (int idx = t; idx < L * COLS; idx += 256) {
        int k1 = idx / COLS, c = idx % COLS;
        float ang = (float)SIGN * TWO_PI * (float)(n2 * k1) / 32768.0f;
        float sw, cw; __sincosf(ang, &sw, &cw);
        float r0 = sre[k1][c], i0 = sim[k1][c];
        size_t dst = (size_t)(k1 * N2 + n2) * 64 + cbase + c;
        tre[dst] = cw * r0 - sw * i0;
        tim[dst] = cw * i0 + sw * r0;
    }
}

template<int SIGN>
__global__ __launch_bounds__(256)
void fft_col_stage2(const float* __restrict__ tre, const float* __restrict__ tim,
                    float* __restrict__ yre, float* __restrict__ yim, float scale)
{
    constexpr int L = 128;
    __shared__ float sre[L][COLS + 1];
    __shared__ float sim[L][COLS + 1];
    const int k1 = blockIdx.x;   // [0,256)
    const int cbase = blockIdx.y * COLS;
    const int t = threadIdx.x;
    for (int idx = t; idx < L * COLS; idx += 256) {
        int n2 = idx / COLS, c = idx % COLS;
        int r = (int)(__brev((unsigned)n2) >> 25);   // 7-bit reverse
        size_t src = (size_t)(k1 * L + n2) * 64 + cbase + c;
        sre[r][c] = tre[src];
        sim[r][c] = tim[src];
    }
    __syncthreads();
    lds_fft<L>(sre, sim, (float)SIGN);
    for (int idx = t; idx < L * COLS; idx += 256) {
        int k2 = idx / COLS, c = idx % COLS;
        size_t dst = (size_t)(k1 + 256 * k2) * 64 + cbase + c;
        yre[dst] = sre[k2][c] * scale;
        yim[dst] = sim[k2][c] * scale;
    }
}

// ---------------------------------------------------------------------------
// Mode mixing: y[b,c] = sum_m crelu(xh[b,c]*w0[m,c]) * phase[m] * w1[m,c]
// Output f16 planes (feeds WMMA lin1 GEMM).
// ---------------------------------------------------------------------------
__global__ __launch_bounds__(256)
void modemix(const float* __restrict__ xre, const float* __restrict__ xim,
             const float* __restrict__ w0,  const float* __restrict__ w1,
             _Float16* __restrict__ ore, _Float16* __restrict__ oim)
{
    __shared__ float sw0[32 * 64], sw1[32 * 64], pr[32], pi[32];
    int t = threadIdx.x;
    for (int i = t; i < 32 * 64; i += 256) { sw0[i] = w0[i]; sw1[i] = w1[i]; }
    if (t < 32) { float a = TWO_PI * (float)t / 32.0f; __sincosf(a, &pi[t], &pr[t]); }
    __syncthreads();
    size_t base = (size_t)blockIdx.x * 256 + t;
    int c = (int)(base & 63);
    float xr = xre[base], xi = xim[base];
    float ar = 0.f, ai = 0.f;
    #pragma unroll
    for (int m = 0; m < 32; ++m) {
        float wm = sw0[m * 64 + c];
        float yr = fmaxf(wm * xr, 0.f), yi = fmaxf(wm * xi, 0.f);
        float zr = yr * pr[m] - yi * pi[m];
        float zi = yr * pi[m] + yi * pr[m];
        float w1m = sw1[m * 64 + c];
        ar += zr * w1m; ai += zi * w1m;
    }
    ore[base] = (_Float16)ar;
    oim[base] = (_Float16)ai;
}

// ---------------------------------------------------------------------------
// M[k,f] = (1/64) sum_w e^{+2pi i k w/64} * W_out[f,w]  (ifft-width fused with W_out)
// ---------------------------------------------------------------------------
__global__ __launch_bounds__(256)
void buildM(const float* __restrict__ Wout, float* __restrict__ Mre, float* __restrict__ Mim)
{
    int idx = blockIdx.x * 256 + threadIdx.x;
    if (idx >= 64 * 128) return;
    int k = idx / 128, f = idx % 128;
    float sr = 0.f, si = 0.f;
    for (int w = 0; w < 64; ++w) {
        float a = TWO_PI * (float)((k * w) & 63) / 64.0f;
        float sw, cw; __sincosf(a, &sw, &cw);
        float v = Wout[f * 64 + w];
        sr += cw * v; si += sw * v;
    }
    Mre[idx] = sr * (1.0f / 64.0f);
    Mim[idx] = si * (1.0f / 64.0f);
}

// ---------------------------------------------------------------------------
// out[b,f] = sum_k y[b,k] * M[k,f] + b_out[f]   (complex, interleaved store)
// ---------------------------------------------------------------------------
__global__ __launch_bounds__(128)
void final_out(const float* __restrict__ yre, const float* __restrict__ yim,
               const float* __restrict__ Mre, const float* __restrict__ Mim,
               const float* __restrict__ bout, float* __restrict__ out)
{
    __shared__ float rr[64], ri[64];
    int b = blockIdx.x, f = threadIdx.x;
    if (f < 64) {
        rr[f] = yre[(size_t)b * 64 + f];
        ri[f] = yim[(size_t)b * 64 + f];
    }
    __syncthreads();
    float sr = bout[f], si = 0.f;
    #pragma unroll 8
    for (int k = 0; k < 64; ++k) {
        float mr = Mre[k * 128 + f], mi = Mim[k * 128 + f];
        sr += rr[k] * mr - ri[k] * mi;
        si += rr[k] * mi + ri[k] * mr;
    }
    size_t o = ((size_t)b * 128 + f) * 2;
    out[o]     = sr;
    out[o + 1] = si;
}

// ---------------------------------------------------------------------------
extern "C" void kernel_launch(void* const* d_in, const int* in_sizes, int n_in,
                              void* d_out, int out_size, void* d_ws, size_t ws_size,
                              hipStream_t stream)
{
    (void)in_sizes; (void)n_in; (void)out_size; (void)ws_size;
    const int B = 32768;

    const float* geom   = (const float*)d_in[0];
    const float* W_in   = (const float*)d_in[1];
    const float* b_in   = (const float*)d_in[2];
    const float* W_h1   = (const float*)d_in[3];
    const float* b_h1   = (const float*)d_in[4];
    const float* W_h2   = (const float*)d_in[5];
    const float* b_h2   = (const float*)d_in[6];
    const float* w0     = (const float*)d_in[7];
    const float* w1     = (const float*)d_in[8];
    const float* lin1_W = (const float*)d_in[9];
    const float* lin1_b = (const float*)d_in[10];
    const float* lin2_W = (const float*)d_in[11];
    const float* lin2_b = (const float*)d_in[12];
    const float* W_out  = (const float*)d_in[13];
    const float* b_out  = (const float*)d_in[14];

    char* ws = (char*)d_ws;
    size_t off = 0;
    auto take = [&](size_t bytes) -> char* {
        char* p = ws + off;
        off += (bytes + 255) & ~(size_t)255;
        return p;
    };

    _Float16* g16    = (_Float16*)take((size_t)B * 512 * 2);
    _Float16* w_in16 = (_Float16*)take((size_t)1024 * 512 * 2);
    _Float16* w_h116 = (_Float16*)take((size_t)512 * 1024 * 2);
    _Float16* w_h216 = (_Float16*)take((size_t)64 * 512 * 2);
    _Float16* l1w16  = (_Float16*)take((size_t)64 * 64 * 2);
    _Float16* l2w16  = (_Float16*)take((size_t)64 * 64 * 2);
    _Float16* x1     = (_Float16*)take((size_t)B * 1024 * 2);
    _Float16* x2     = (_Float16*)take((size_t)B * 512 * 2);
    float*    x3     = (float*)take((size_t)B * 64 * 4);
    float*    pAre   = (float*)take((size_t)B * 64 * 4);
    float*    pAim   = (float*)take((size_t)B * 64 * 4);
    float*    pBre   = (float*)take((size_t)B * 64 * 4);
    float*    pBim   = (float*)take((size_t)B * 64 * 4);
    _Float16* h16re  = (_Float16*)take((size_t)B * 64 * 2);
    _Float16* h16im  = (_Float16*)take((size_t)B * 64 * 2);
    _Float16* l16re  = (_Float16*)take((size_t)B * 64 * 2);
    _Float16* l16im  = (_Float16*)take((size_t)B * 64 * 2);
    float*    Mre    = (float*)take((size_t)64 * 128 * 4);
    float*    Mim    = (float*)take((size_t)64 * 128 * 4);
    float*    zeros  = (float*)take(256 * 4);

    (void)hipMemsetAsync(zeros, 0, 256 * sizeof(float), stream);

    auto cvt = [&](const float* src, _Float16* dst, int n) {
        cvt_f32_f16<<<(n + 255) / 256, 256, 0, stream>>>(src, dst, n);
    };
    cvt(geom,   g16,    B * 512);
    cvt(W_in,   w_in16, 1024 * 512);
    cvt(W_h1,   w_h116, 512 * 1024);
    cvt(W_h2,   w_h216, 64 * 512);
    cvt(lin1_W, l1w16,  64 * 64);
    cvt(lin2_W, l2w16,  64 * 64);

    // MLP encoder (WMMA f16, f32 accumulate, TDM-staged operands)
    gemm_f16<4, 2, true, true ><<<dim3(B / 128, 1024 / 128), 256, 0, stream>>>(
        g16, w_in16, b_in, x1, B, 1024, 512);
    gemm_f16<4, 2, true, true ><<<dim3(B / 128, 512 / 128), 256, 0, stream>>>(
        x1, w_h116, b_h1, x2, B, 512, 1024);
    gemm_f16<4, 1, true, false><<<dim3(B / 128, 1), 128, 0, stream>>>(
        x2, w_h216, b_h2, x3, B, 64, 512);

    // forward fftn: width-64 DFT then 32768-pt batch FFT (four-step)
    fft_width64<<<B / 4, 256, 0, stream>>>(x3, pAre, pAim);
    fft_col_stage1<-1><<<dim3(128, 4), 256, 0, stream>>>(pAre, pAim, pBre, pBim);
    fft_col_stage2<-1><<<dim3(256, 4), 256, 0, stream>>>(pBre, pBim, pAre, pAim, 1.0f);

    // mode mixing -> f16 planes
    modemix<<<(B * 64) / 256, 256, 0, stream>>>(pAre, pAim, w0, w1, h16re, h16im);

    // lin1 (crelu) / lin2 on re & im planes via WMMA
    gemm_f16<4, 1, true, true ><<<dim3(B / 128, 1), 128, 0, stream>>>(
        h16re, l1w16, lin1_b, l16re, B, 64, 64);
    gemm_f16<4, 1, true, true ><<<dim3(B / 128, 1), 128, 0, stream>>>(
        h16im, l1w16, zeros, l16im, B, 64, 64);
    gemm_f16<4, 1, false, false><<<dim3(B / 128, 1), 128, 0, stream>>>(
        l16re, l2w16, lin2_b, pAre, B, 64, 64);
    gemm_f16<4, 1, false, false><<<dim3(B / 128, 1), 128, 0, stream>>>(
        l16im, l2w16, zeros, pAim, B, 64, 64);

    // inverse batch-axis FFT (scale 1/32768 here; 1/64 folded into M)
    fft_col_stage1<+1><<<dim3(128, 4), 256, 0, stream>>>(pAre, pAim, pBre, pBim);
    fft_col_stage2<+1><<<dim3(256, 4), 256, 0, stream>>>(pBre, pBim, pAre, pAim,
                                                         1.0f / 32768.0f);

    // fused (ifft-width ∘ W_out) complex matrix, then final projection
    buildM<<<(64 * 128 + 255) / 256, 256, 0, stream>>>(W_out, Mre, Mim);
    final_out<<<B, 128, 0, stream>>>(pAre, pAim, Mre, Mim, b_out, (float*)d_out);
}
